// GateAttention_62826781606062
// MI455X (gfx1250) — matrix operand order repaired and checked by reading, not compiled, not run on hardware
//
#include <hip/hip_runtime.h>
#include <hip/hip_bf16.h>

// ---------------------------------------------------------------------------
// GAU gated attention for MI455X (gfx1250), bf16 WMMA path, M=32 blocks.
//   B=8, L=2048, E=128, DV=1024
//   out = u * ( (relu(q·k^T / sqrt(E))^2 masked-causal / (m+1)) · v )
// Each block: 32 query rows; 8 wave32; wave w owns DV slice [128w,128w+128)
// and two M-halves -> each v B-operand feeds TWO wmma (2x reuse vs M=16).
// ---------------------------------------------------------------------------

#define GAU_B  8
#define GAU_L  2048
#define GAU_E  128
#define GAU_DV 1024

typedef __attribute__((ext_vector_type(8)))  float  v8f;
typedef __attribute__((ext_vector_type(16))) __bf16 v16bf;

union bfop { v16bf v; uint4 q[2]; };

static __device__ __forceinline__ unsigned short f32_to_bf16(float f) {
    union { float f; unsigned int u; } t; t.f = f;
    unsigned int r = t.u + (0x7FFFu + ((t.u >> 16) & 1u));   // RNE
    return (unsigned short)(r >> 16);
}

// -------------------- preprocessing: f32 -> bf16 (row major) ----------------
__global__ void cvt_bf16_kernel(const float* __restrict__ src,
                                unsigned short* __restrict__ dst, int n) {
    int i = blockIdx.x * blockDim.x + threadIdx.x;
    if (i < n) dst[i] = f32_to_bf16(src[i]);
}

// -------------------- preprocessing: v -> vT bf16 [B][DV][L] ----------------
__global__ void transpose_v_kernel(const float* __restrict__ v,
                                   unsigned short* __restrict__ vT) {
    __shared__ float tile[32][33];
    const int b  = blockIdx.z;
    const int l0 = blockIdx.x * 32;
    const int d0 = blockIdx.y * 32;
    const int tx = threadIdx.x, ty = threadIdx.y;      // 32 x 8
#pragma unroll
    for (int i = 0; i < 32; i += 8)
        tile[ty + i][tx] = v[((size_t)b * GAU_L + (l0 + ty + i)) * GAU_DV + d0 + tx];
    __syncthreads();
#pragma unroll
    for (int i = 0; i < 32; i += 8)
        vT[((size_t)b * GAU_DV + (d0 + ty + i)) * GAU_L + l0 + tx] =
            f32_to_bf16(tile[tx][ty + i]);
}

// -------------------- main fused GAU attention kernel -----------------------
// grid: (L/32, B), block: 256 threads (8 wave32)
__global__ __launch_bounds__(256)
void gau_attn_kernel(const float* __restrict__ u,
                     const unsigned short* __restrict__ qb,
                     const unsigned short* __restrict__ kb,
                     const unsigned short* __restrict__ vT,
                     float* __restrict__ out) {
    // shared bf16 attention-weight tile: 32 query rows x 128 keys
    __shared__ __align__(16) unsigned short sA[32 * 128];

    const int lane = threadIdx.x & 31;
    const int wave = threadIdx.x >> 5;
    const int half = lane >> 4;          // 0: lanes 0-15, 1: lanes 16-31
    const int lid  = lane & 15;
    const int m0   = blockIdx.x * 32;
    const int b    = blockIdx.y;
    const float scale = 0.088388347648318447f;  // 1/sqrt(128)

    // score-tile assignment for this wave: M-half mt, key pair kp
    const int mt = wave >> 2;            // 0 or 1  (query rows m0+16mt .. +16)
    const int kp = wave & 3;             // keys [n0+32kp, n0+32kp+32)

    v8f acc[2][8] = {};                  // 32 x 128 f32 accumulator slice
    const int mMax   = m0 + 31;
    const int dvbase = wave * 128;

    for (int n0 = 0; n0 <= mMax; n0 += 128) {
        // ---- scores: this wave computes 2 16x16 S tiles (32 keys) ----------
        v8f s0 = {}, s1 = {};
        {
            const unsigned short* qrow =
                qb + ((size_t)(b * GAU_L + m0 + 16 * mt + lid) * GAU_E);
            const int key0 = n0 + 32 * kp + lid;
            const unsigned short* krow0 =
                kb + ((size_t)(b * GAU_L + key0) * GAU_E);
            const unsigned short* krow1 = krow0 + (size_t)16 * GAU_E;
#pragma unroll
            for (int c = 0; c < 4; ++c) {
                // A (16x32): lanes0-15 K {c32+0..7, c32+16..23}, lanes16-31 +8
                const int ab = c * 32 + half * 8;
                bfop ta;
                ta.q[0] = *(const uint4*)(qrow + ab);
                ta.q[1] = *(const uint4*)(qrow + ab + 16);
                // B (32x16): lanes0-15 K {c32+0..7, c32+16..23}, lanes16-31 +8
                const int eb = c * 32 + half * 8;
                bfop tk;
                tk.q[0] = *(const uint4*)(krow0 + eb);
                tk.q[1] = *(const uint4*)(krow0 + eb + 16);
                s0 = __builtin_amdgcn_wmma_f32_16x16x32_bf16(
                         false, ta.v, false, tk.v, (short)0, s0, false, false);
                tk.q[0] = *(const uint4*)(krow1 + eb);
                tk.q[1] = *(const uint4*)(krow1 + eb + 16);
                s1 = __builtin_amdgcn_wmma_f32_16x16x32_bf16(
                         false, ta.v, false, tk.v, (short)0, s1, false, false);
            }
        }

        // prefetch next key tile's k rows while we post-process
        if (n0 + 128 <= mMax) {
            __builtin_prefetch(
                kb + ((size_t)(b * GAU_L + n0 + 128 + 32 * kp + lid) * GAU_E),
                0, 1);
        }

        // ---- mask + squared-relu normalize, write bf16 A-image to LDS ------
        {
            const int n_abs0 = n0 + 32 * kp + lid;      // subtile 0 key col
            const int rowb   = 16 * mt + half * 8;
#pragma unroll
            for (int j = 0; j < 8; ++j) {
                const int m_abs = m0 + rowb + j;        // C/D layout row
                const float inv = 1.0f / (float)(m_abs + 1);
                float x0 = fmaxf(s0[j] * scale, 0.0f);
                x0 = x0 * x0 * inv;
                x0 = (n_abs0 <= m_abs) ? x0 : 0.0f;
                float x1 = fmaxf(s1[j] * scale, 0.0f);
                x1 = x1 * x1 * inv;
                x1 = (n_abs0 + 16 <= m_abs) ? x1 : 0.0f;
                unsigned short* dst = sA + (rowb + j) * 128 + 32 * kp + lid;
                dst[0]  = f32_to_bf16(x0);
                dst[16] = f32_to_bf16(x1);
            }
        }
        __syncthreads();

        // ---- A·v over this 128-key tile (K = 128 -> 4 chunks of 32) --------
        // Each v B-operand is reused by BOTH M-halves (2 wmma per load).
#pragma unroll
        for (int c = 0; c < 4; ++c) {
            const int ab = c * 32 + half * 8;
            bfop ta0, ta1;
            {
                const unsigned short* ar0 = sA + lid * 128 + ab;
                const unsigned short* ar1 = sA + (16 + lid) * 128 + ab;
                ta0.q[0] = *(const uint4*)(ar0);
                ta0.q[1] = *(const uint4*)(ar0 + 16);
                ta1.q[0] = *(const uint4*)(ar1);
                ta1.q[1] = *(const uint4*)(ar1 + 16);
            }
            const unsigned short* vbp =
                vT + ((size_t)(b * GAU_DV + dvbase + lid) * GAU_L)
                   + n0 + c * 32 + half * 8;
#pragma unroll
            for (int d = 0; d < 8; ++d) {
                bfop tb;
                const unsigned short* vrow = vbp + (size_t)d * 16 * GAU_L;
                tb.q[0] = *(const uint4*)(vrow);
                tb.q[1] = *(const uint4*)(vrow + 16);
                acc[0][d] = __builtin_amdgcn_wmma_f32_16x16x32_bf16(
                                false, ta0.v, false, tb.v, (short)0,
                                acc[0][d], false, false);
                acc[1][d] = __builtin_amdgcn_wmma_f32_16x16x32_bf16(
                                false, ta1.v, false, tb.v, (short)0,
                                acc[1][d], false, false);
            }
        }
        __syncthreads();   // protect sA before next tile overwrites it
    }

    // ---- gate by u and store f32 ------------------------------------------
#pragma unroll
    for (int t = 0; t < 2; ++t) {
#pragma unroll
        for (int d = 0; d < 8; ++d) {
            const int n_dv = dvbase + d * 16 + lid;
#pragma unroll
            for (int j = 0; j < 8; ++j) {
                const int m_abs = m0 + 16 * t + j + half * 8;
                const size_t idx =
                    ((size_t)(b * GAU_L + m_abs)) * GAU_DV + n_dv;
                out[idx] = acc[t][d][j] * u[idx];
            }
        }
    }
}

// ---------------------------------------------------------------------------
extern "C" void kernel_launch(void* const* d_in, const int* in_sizes, int n_in,
                              void* d_out, int out_size, void* d_ws,
                              size_t ws_size, hipStream_t stream) {
    (void)in_sizes; (void)n_in; (void)out_size; (void)ws_size;

    const float* u = (const float*)d_in[0];
    const float* q = (const float*)d_in[1];
    const float* k = (const float*)d_in[2];
    const float* v = (const float*)d_in[3];
    // d_in[4] = attn_mask: implicit (causal), not read.
    float* out = (float*)d_out;

    // workspace layout (bf16): qb | kb | vT   (4 MiB + 4 MiB + 32 MiB)
    const int nQK = GAU_B * GAU_L * GAU_E;          // 2,097,152
    unsigned short* qb = (unsigned short*)d_ws;
    unsigned short* kb = qb + nQK;
    unsigned short* vT = kb + nQK;

    cvt_bf16_kernel<<<(nQK + 255) / 256, 256, 0, stream>>>(q, qb, nQK);
    cvt_bf16_kernel<<<(nQK + 255) / 256, 256, 0, stream>>>(k, kb, nQK);

    dim3 tg(GAU_L / 32, GAU_DV / 32, GAU_B);        // 64 x 32 x 8
    transpose_v_kernel<<<tg, dim3(32, 8), 0, stream>>>(v, vT);

    dim3 grid(GAU_L / 32, GAU_B);                   // 64 x 8
    gau_attn_kernel<<<grid, 256, 0, stream>>>(u, qb, kb, vT, out);
}
